// CSOCSSC_v114_45346264711616
// MI455X (gfx1250) — matrix operand order repaired and checked by reading, not compiled, not run on hardware
//
#include <hip/hip_runtime.h>
#include <hip/hip_bf16.h>
#include <math.h>

// ---------------------------------------------------------------------------
// Model constants (from reference): D=256, H=8, L=4, N=512, BINS=64, DFF=1024
// ---------------------------------------------------------------------------
#define NSEQ 512
#define DMODEL 256
#define NHEAD 8
#define DHEAD 32
#define NLAYER 4
#define DFF 1024
#define NBINS 64

typedef __attribute__((ext_vector_type(16))) _Float16 v16h;
typedef __attribute__((ext_vector_type(8)))  _Float16 v8h;
typedef __attribute__((ext_vector_type(8)))  float    v8f;

// ---------------------------------------------------------------------------
// Device helpers
// ---------------------------------------------------------------------------
__device__ __forceinline__ float gelu_exact(float x) {
    return 0.5f * x * (1.0f + erff(x * 0.70710678118654752f));
}

__device__ __forceinline__ float wave_sum(float v) {
#pragma unroll
    for (int o = 16; o > 0; o >>= 1) v += __shfl_xor(v, o, 32);
    return v;
}

__device__ __forceinline__ float wave_max(float v) {
#pragma unroll
    for (int o = 16; o > 0; o >>= 1) v = fmaxf(v, __shfl_xor(v, o, 32));
    return v;
}

// blockDim.x == 256
__device__ __forceinline__ float block_sum256(float v, float* red) {
    int t = threadIdx.x;
    red[t] = v; __syncthreads();
#pragma unroll
    for (int s = 128; s > 0; s >>= 1) {
        if (t < s) red[t] += red[t + s];
        __syncthreads();
    }
    float r = red[0]; __syncthreads();
    return r;
}

// Pack 8 f32 -> 8 f16 lanes of a v16h starting at BASE (ISA A-fragment layout).
template <int BASE>
__device__ __forceinline__ void pack8(v16h& d, const float4& f0, const float4& f1) {
    d[BASE + 0] = (_Float16)f0.x; d[BASE + 1] = (_Float16)f0.y;
    d[BASE + 2] = (_Float16)f0.z; d[BASE + 3] = (_Float16)f0.w;
    d[BASE + 4] = (_Float16)f1.x; d[BASE + 5] = (_Float16)f1.y;
    d[BASE + 6] = (_Float16)f1.z; d[BASE + 7] = (_Float16)f1.w;
}

// ---------------------------------------------------------------------------
// Generic WMMA GEMM:  C[M,Nout] = act( A[M,K] * W[Nout,K]^T + bias )
// A row-major (lda), W row-major [out,in] (ldw) -> computes X @ W^T directly.
// One wave per 16x16 C tile; block = 4 waves = 4 consecutive N-tiles.
// Grid: (M/16, Nout/64), block 128.   ACT: 0 none, 1 relu, 2 gelu, 3 sigmoid
// ---------------------------------------------------------------------------
template <int ACT>
__global__ void gemm_wmma_kernel(const float* __restrict__ A, int lda,
                                 const float* __restrict__ W, int ldw,
                                 const float* __restrict__ bias,
                                 float* __restrict__ C, int ldc, int K) {
    const int wave = threadIdx.x >> 5;
    const int lane = threadIdx.x & 31;
    const int r    = lane & 15;     // row-in-tile (A) / col-in-tile (B,C)
    const int hf   = lane >> 4;     // K-group selector per ISA layout
    const int tm   = blockIdx.x;
    const int tn   = blockIdx.y * 4 + wave;

    const float* Ar = A + (size_t)(tm * 16 + r) * lda;
    const float* Wr = W + (size_t)(tn * 16 + r) * ldw;

    v8f acc;
    const float bv = bias ? bias[tn * 16 + r] : 0.0f;
#pragma unroll
    for (int v = 0; v < 8; ++v) acc[v] = bv;

    for (int k0 = 0; k0 < K; k0 += 32) {
        v16h a, b;
        // A fragment: lane(r,hf) holds A[row, k0+hf*8 ..+7] and A[row, k0+16+hf*8 ..+7]
        const float4* pa0 = (const float4*)(Ar + k0 + hf * 8);
        const float4* pa1 = (const float4*)(Ar + k0 + 16 + hf * 8);
        pack8<0>(a, pa0[0], pa0[1]);
        pack8<8>(a, pa1[0], pa1[1]);
        // B fragment: lane(r,hf) holds W[col, k0+hf*8 ..+7] and W[col, k0+16+hf*8 ..+7]
        const float4* pw0 = (const float4*)(Wr + k0 + hf * 8);
        const float4* pw1 = (const float4*)(Wr + k0 + 16 + hf * 8);
        float4 w0 = pw0[0], w1 = pw0[1], w2 = pw1[0], w3 = pw1[1];
        pack8<0>(b, w0, w1);
        pack8<8>(b, w2, w3);

        acc = __builtin_amdgcn_wmma_f32_16x16x32_f16(
            /*neg_a=*/false, a, /*neg_b=*/false, b,
            /*c_mod=*/(short)0, acc, /*reuse_a=*/false, /*reuse_b=*/false);
    }

    // C fragment: lane covers col = tn*16+r, rows tm*16 + hf*8 + v
    float* Cr = C + (size_t)(tm * 16 + hf * 8) * ldc + tn * 16 + r;
#pragma unroll
    for (int v = 0; v < 8; ++v) {
        float val = acc[v];
        if (ACT == 1)      val = fmaxf(val, 0.0f);
        else if (ACT == 2) val = gelu_exact(val);
        else if (ACT == 3) val = 1.0f / (1.0f + expf(-val));
        Cr[(size_t)v * ldc] = val;
    }
}

// ---------------------------------------------------------------------------
// f32 -> f16 conversion pre-pass (used once for dW2).  grid(ceil(n/256)), b(256)
// ---------------------------------------------------------------------------
__global__ void cvt_f16_kernel(const float* __restrict__ src,
                               _Float16* __restrict__ dst, int n) {
    const int i = blockIdx.x * 256 + threadIdx.x;
    if (i < n) dst[i] = (_Float16)src[i];
}

// ---------------------------------------------------------------------------
// Embedding + sinusoidal positional encoding.  grid(N), block(D)
// ---------------------------------------------------------------------------
__global__ void embed_kernel(const int* __restrict__ tokens,
                             const float* __restrict__ emb,
                             float* __restrict__ x) {
    const int n = blockIdx.x, d = threadIdx.x;
    const float e = emb[tokens[n] * DMODEL + d];
    const int i2 = d & ~1;
    const float div = expf(-(float)i2 * (9.210340371976184f / (float)DMODEL));
    const float arg = (float)n * div;
    const float pe = (d & 1) ? cosf(arg) : sinf(arg);
    x[n * DMODEL + d] = e + pe;
}

// ---------------------------------------------------------------------------
// Attention core for one (query i, head h): one wave. DHEAD==32==wave size.
// grid(N, H), block(32)
// ---------------------------------------------------------------------------
__global__ void attn_kernel(const float* __restrict__ qkv, float* __restrict__ o) {
    __shared__ float sc[NSEQ];
    const int i = blockIdx.x, h = blockIdx.y, lane = threadIdx.x;
    const float scale = 0.17677669529663687f;  // 1/sqrt(32)

    const float* q = qkv + (size_t)i * 3 * DMODEL + h * DHEAD;
    float lmax = -3.4e38f;
    for (int j = lane; j < NSEQ; j += 32) {
        const float* k = qkv + (size_t)j * 3 * DMODEL + DMODEL + h * DHEAD;
        float s = 0.0f;
#pragma unroll
        for (int d = 0; d < DHEAD; ++d) s += q[d] * k[d];
        s *= scale;
        sc[j] = s;
        lmax = fmaxf(lmax, s);
    }
    lmax = wave_max(lmax);
    float lsum = 0.0f;
    for (int j = lane; j < NSEQ; j += 32) {
        float e = expf(sc[j] - lmax);
        sc[j] = e;
        lsum += e;
    }
    lsum = wave_sum(lsum);
    const float inv = 1.0f / lsum;
    __syncthreads();

    const int d = lane;  // one output dim per lane
    float acc = 0.0f;
    for (int j = 0; j < NSEQ; ++j)
        acc += sc[j] * qkv[(size_t)j * 3 * DMODEL + 2 * DMODEL + h * DHEAD + d];
    o[(size_t)i * DMODEL + h * DHEAD + d] = acc * inv;
}

// ---------------------------------------------------------------------------
// out = LayerNorm(res + delta) * g + b.  grid(N), block(D=256). In-place safe.
// ---------------------------------------------------------------------------
__global__ void add_ln_kernel(const float* __restrict__ res,
                              const float* __restrict__ delta,
                              const float* __restrict__ g,
                              const float* __restrict__ b,
                              float* __restrict__ out) {
    __shared__ float red[DMODEL];
    const int n = blockIdx.x, d = threadIdx.x;
    const float v = res[n * DMODEL + d] + delta[n * DMODEL + d];
    const float mean = block_sum256(v, red) * (1.0f / DMODEL);
    const float c = v - mean;
    const float var = block_sum256(c * c, red) * (1.0f / DMODEL);
    out[n * DMODEL + d] = c * rsqrtf(var + 1e-5f) * g[d] + b[d];
}

// ---------------------------------------------------------------------------
// alpha[i] = 1.2 + 2.3 * sigmoid(a1[i,:] . aW2 + ab2).  grid(N), block(32)
// ---------------------------------------------------------------------------
__global__ void alpha_kernel(const float* __restrict__ a1,
                             const float* __restrict__ aW2,
                             const float* __restrict__ ab2,
                             float* __restrict__ alpha) {
    const int i = blockIdx.x, lane = threadIdx.x;
    float s = 0.0f;
    for (int k = lane; k < DMODEL; k += 32) s += a1[i * DMODEL + k] * aW2[k];
    s = wave_sum(s);
    if (lane == 0) {
        const float sig = 1.0f / (1.0f + expf(-(s + ab2[0])));
        alpha[i] = 1.2f + sig * 2.3f;
    }
}

// ---------------------------------------------------------------------------
// CSOC kernel attention row: build power-law*exp kernel row, normalize,
// latent[i,:] = kn . x.  grid(N), block(D=256)
// ---------------------------------------------------------------------------
__global__ void csoc_kernel(const float* __restrict__ x,
                            const float* __restrict__ alpha,
                            float* __restrict__ latent) {
    __shared__ float w[NSEQ];
    __shared__ float red[DMODEL];
    const int i = blockIdx.x, t = threadIdx.x;
    const float ai = alpha[i];
    float part = 0.0f;
    for (int j = t; j < NSEQ; j += DMODEL) {
        const float r = fabsf((float)(i - j)) + 1e-4f;
        const float ap = 0.5f * (ai + alpha[j]);
        const float kv = expf(-ap * logf(r) - r * 0.1f);  // r^-ap * exp(-r/10)
        w[j] = kv;
        part += kv;
    }
    const float sum = block_sum256(part, red);
    const float inv = 1.0f / (sum + 1e-8f);
    float acc = 0.0f;
    for (int j = 0; j < NSEQ; ++j) acc += w[j] * x[(size_t)j * DMODEL + t];
    latent[(size_t)i * DMODEL + t] = acc * inv;
}

// ---------------------------------------------------------------------------
// Distogram tile kernel.  disto[i, j, b] = gelu(U[i]+V[j]) . dW2[b,:] + db2[b]
// Block = 4 waves; fixed i and 16-row j tile.
//  - A-tile (16 x 256, f16) built once in LDS, shared by all 4 waves.
//  - B read directly as pre-converted f16 (no inner-loop cvt).
//  - C staged through LDS and stored as a fully contiguous 4KB block of
//    float4s (this stream is the HBM roofline term: 64 MB total).
// grid(N, N/16), block(128)
// ---------------------------------------------------------------------------
__global__ void disto_kernel(const float* __restrict__ U,
                             const float* __restrict__ V,
                             const _Float16* __restrict__ W2h,
                             const float* __restrict__ db2,
                             float* __restrict__ out) {
    __shared__ __align__(16) _Float16 Hs[16][DMODEL + 16];
    __shared__ float Cs[16][NBINS + 4];
    const int i = blockIdx.x, jt = blockIdx.y, t = threadIdx.x;

    // Phase 1: H = gelu(U[i] + V[j]) for 16 j rows, f32 -> f16 into LDS.
    for (int idx = t; idx < 16 * DMODEL; idx += 128) {
        const int jr = idx >> 8;
        const int d  = idx & (DMODEL - 1);
        const float hv = U[(size_t)i * DMODEL + d] +
                         V[(size_t)(jt * 16 + jr) * DMODEL + d];
        Hs[jr][d] = (_Float16)gelu_exact(hv);
    }
    __syncthreads();

    // Phase 2: each wave computes one 16(j) x 16(bin) tile via WMMA.
    const int wave = t >> 5, lane = t & 31;
    const int r = lane & 15, hf = lane >> 4;
    const int brow = wave * 16 + r;

    v8f acc;
    const float bv = db2[brow];
#pragma unroll
    for (int v = 0; v < 8; ++v) acc[v] = bv;

    const _Float16* Wr = W2h + (size_t)brow * DMODEL;
    for (int k0 = 0; k0 < DMODEL; k0 += 32) {
        v16h a, b;
        const v8h a0 = *(const v8h*)&Hs[r][k0 + hf * 8];
        const v8h a1 = *(const v8h*)&Hs[r][k0 + 16 + hf * 8];
        const v8h b0 = *(const v8h*)(Wr + k0 + hf * 8);
        const v8h b1 = *(const v8h*)(Wr + k0 + 16 + hf * 8);
#pragma unroll
        for (int q = 0; q < 8; ++q) {
            a[q] = a0[q]; a[8 + q] = a1[q];
            b[q] = b0[q]; b[8 + q] = b1[q];
        }
        acc = __builtin_amdgcn_wmma_f32_16x16x32_f16(
            false, a, false, b, (short)0, acc, false, false);
    }

    // Phase 3: stage C tile in LDS (bank-conflict-free: stride 68 = 4 mod 64),
    // then emit the block's contiguous 4KB output region as coalesced float4s.
#pragma unroll
    for (int v = 0; v < 8; ++v) Cs[hf * 8 + v][wave * 16 + r] = acc[v];
    __syncthreads();

    float* obase = out + (size_t)i * NSEQ * NBINS + (size_t)jt * 16 * NBINS;
    for (int idx = t * 4; idx < 16 * NBINS; idx += 128 * 4) {
        const int jr = idx >> 6;
        const int b0i = idx & (NBINS - 1);
        float4 val;
        val.x = Cs[jr][b0i + 0];
        val.y = Cs[jr][b0i + 1];
        val.z = Cs[jr][b0i + 2];
        val.w = Cs[jr][b0i + 3];
        *(float4*)(obase + idx) = val;
    }
}

// ---------------------------------------------------------------------------
// Torsion head tail: phi/psi = tanh(t1 . tW2^T + tb2) * pi.  grid(N), block(32)
// ---------------------------------------------------------------------------
__global__ void torsion_kernel(const float* __restrict__ t1,
                               const float* __restrict__ tW2,
                               const float* __restrict__ tb2,
                               float* __restrict__ phi,
                               float* __restrict__ psi) {
    const int i = blockIdx.x, lane = threadIdx.x;
    float s0 = 0.0f, s1 = 0.0f;
    for (int k = lane; k < DMODEL; k += 32) {
        const float x = t1[i * DMODEL + k];
        s0 += x * tW2[k];
        s1 += x * tW2[DMODEL + k];
    }
    s0 = wave_sum(s0); s1 = wave_sum(s1);
    if (lane == 0) {
        phi[i] = tanhf(s0 + tb2[0]) * 3.14159265358979323846f;
        psi[i] = tanhf(s1 + tb2[1]) * 3.14159265358979323846f;
    }
}

// ---------------------------------------------------------------------------
// Coordinate head: target = latent . pW^T + pb; unrolled diffusion fold.
// grid(N), block(32)
// ---------------------------------------------------------------------------
__global__ void coords_kernel(const float* __restrict__ latent,
                              const float* __restrict__ pW,
                              const float* __restrict__ pb,
                              const float* __restrict__ noise,
                              float* __restrict__ cout) {
    const int i = blockIdx.x, lane = threadIdx.x;
    float s0 = 0.0f, s1 = 0.0f, s2 = 0.0f;
    for (int k = lane; k < DMODEL; k += 32) {
        const float x = latent[i * DMODEL + k];
        s0 += x * pW[k];
        s1 += x * pW[DMODEL + k];
        s2 += x * pW[2 * DMODEL + k];
    }
    s0 = wave_sum(s0); s1 = wave_sum(s1); s2 = wave_sum(s2);
    if (lane < 3) {
        const float tgt = ((lane == 0) ? s0 : (lane == 1) ? s1 : s2) + pb[lane];
        float c = noise[i * 3 + lane];
#pragma unroll
        for (int step = 9; step >= 0; --step) {
            const float a = (float)(step + 1) * 0.1f;
            c = a * c + (1.0f - a) * tgt;
        }
        cout[i * 3 + lane] = c;
    }
}

// ---------------------------------------------------------------------------
// Host-side GEMM dispatcher
// ---------------------------------------------------------------------------
static inline void run_gemm(int act, const float* A, int lda, const float* W, int ldw,
                            const float* bias, float* C, int ldc,
                            int M, int Nout, int K, hipStream_t s) {
    dim3 g(M / 16, Nout / 64), b(128);
    switch (act) {
        case 1: gemm_wmma_kernel<1><<<g, b, 0, s>>>(A, lda, W, ldw, bias, C, ldc, K); break;
        case 2: gemm_wmma_kernel<2><<<g, b, 0, s>>>(A, lda, W, ldw, bias, C, ldc, K); break;
        case 3: gemm_wmma_kernel<3><<<g, b, 0, s>>>(A, lda, W, ldw, bias, C, ldc, K); break;
        default: gemm_wmma_kernel<0><<<g, b, 0, s>>>(A, lda, W, ldw, bias, C, ldc, K); break;
    }
}

extern "C" void kernel_launch(void* const* d_in, const int* in_sizes, int n_in,
                              void* d_out, int out_size, void* d_ws, size_t ws_size,
                              hipStream_t stream) {
    (void)in_sizes; (void)n_in; (void)out_size; (void)ws_size;

    // ---- inputs (setup_inputs order) ----
    const int*   tokens = (const int*)d_in[0];
    const float* noise  = (const float*)d_in[1];
    const float* emb    = (const float*)d_in[2];
    const float* Wqkv   = (const float*)d_in[3];
    const float* bqkv   = (const float*)d_in[4];
    const float* Wo     = (const float*)d_in[5];
    const float* bo     = (const float*)d_in[6];
    const float* ln1_g  = (const float*)d_in[7];
    const float* ln1_b  = (const float*)d_in[8];
    const float* W1     = (const float*)d_in[9];
    const float* b1     = (const float*)d_in[10];
    const float* W2     = (const float*)d_in[11];
    const float* b2     = (const float*)d_in[12];
    const float* ln2_g  = (const float*)d_in[13];
    const float* ln2_b  = (const float*)d_in[14];
    const float* aW1    = (const float*)d_in[15];
    const float* ab1    = (const float*)d_in[16];
    const float* aW2    = (const float*)d_in[17];
    const float* ab2    = (const float*)d_in[18];
    const float* cW     = (const float*)d_in[19];
    const float* cb     = (const float*)d_in[20];
    const float* dW1    = (const float*)d_in[21];
    const float* db1    = (const float*)d_in[22];
    const float* dW2    = (const float*)d_in[23];
    const float* db2    = (const float*)d_in[24];
    const float* tW1    = (const float*)d_in[25];
    const float* tb1    = (const float*)d_in[26];
    const float* tW2    = (const float*)d_in[27];
    const float* tb2    = (const float*)d_in[28];
    const float* pW     = (const float*)d_in[29];
    const float* pb     = (const float*)d_in[30];

    // ---- workspace carve (floats), ~6.9 MB total ----
    float* ws   = (float*)d_ws;
    float* x    = ws;                 // [512,256]
    float* qkv  = ws + 131072;        // [512,768]
    float* obuf = ws + 524288;        // [512,256]
    float* tmp  = ws + 655360;        // [512,256]
    float* ff   = ws + 786432;        // [512,1024]
    float* Ub   = ws + 1310720;       // [512,256]
    float* Vb   = ws + 1441792;       // [512,256]
    float* buf3 = ws + 1572864;       // [512,256]
    _Float16* W2h = (_Float16*)(ws + 1703936);  // [64,256] f16 (32 KB)

    // ---- outputs (flat concat, return order) ----
    float* out       = (float*)d_out;
    float* latent_o  = out;                // 512*256
    float* alpha_o   = out + 131072;       // 512
    float* contact_o = out + 131584;       // 512*512
    float* disto_o   = out + 393728;       // 512*512*64
    float* phi_o     = out + 17170944;     // 512
    float* psi_o     = out + 17171456;     // 512
    float* c_o       = out + 17171968;     // 512*3

    // ---- embedding + positional encoding ----
    embed_kernel<<<dim3(NSEQ), dim3(DMODEL), 0, stream>>>(tokens, emb, x);

    // ---- pre-convert dW2 to f16 (tiny; removes cvt from distogram hot loop) ----
    cvt_f16_kernel<<<dim3((NBINS * DMODEL + 255) / 256), dim3(256), 0, stream>>>(
        dW2, W2h, NBINS * DMODEL);

    // ---- transformer trunk ----
    for (int l = 0; l < NLAYER; ++l) {
        const float* Wq  = Wqkv + (size_t)l * 3 * DMODEL * DMODEL;
        const float* bq  = bqkv + (size_t)l * 3 * DMODEL;
        const float* Wol = Wo   + (size_t)l * DMODEL * DMODEL;
        const float* bol = bo   + (size_t)l * DMODEL;
        const float* g1  = ln1_g + (size_t)l * DMODEL;
        const float* be1 = ln1_b + (size_t)l * DMODEL;
        const float* W1l = W1 + (size_t)l * DFF * DMODEL;
        const float* b1l = b1 + (size_t)l * DFF;
        const float* W2l = W2 + (size_t)l * DMODEL * DFF;
        const float* b2l = b2 + (size_t)l * DMODEL;
        const float* g2  = ln2_g + (size_t)l * DMODEL;
        const float* be2 = ln2_b + (size_t)l * DMODEL;

        run_gemm(0, x, DMODEL, Wq, DMODEL, bq, qkv, 3 * DMODEL,
                 NSEQ, 3 * DMODEL, DMODEL, stream);
        attn_kernel<<<dim3(NSEQ, NHEAD), dim3(32), 0, stream>>>(qkv, obuf);
        run_gemm(0, obuf, DMODEL, Wol, DMODEL, bol, tmp, DMODEL,
                 NSEQ, DMODEL, DMODEL, stream);
        add_ln_kernel<<<dim3(NSEQ), dim3(DMODEL), 0, stream>>>(x, tmp, g1, be1, x);
        run_gemm(1, x, DMODEL, W1l, DMODEL, b1l, ff, DFF,
                 NSEQ, DFF, DMODEL, stream);
        run_gemm(0, ff, DFF, W2l, DFF, b2l, tmp, DMODEL,
                 NSEQ, DMODEL, DFF, stream);
        add_ln_kernel<<<dim3(NSEQ), dim3(DMODEL), 0, stream>>>(x, tmp, g2, be2, x);
    }

    // ---- AdaptiveAlphaPredictor ----
    run_gemm(2, x, DMODEL, aW1, DMODEL, ab1, buf3, DMODEL,
             NSEQ, DMODEL, DMODEL, stream);
    alpha_kernel<<<dim3(NSEQ), dim3(32), 0, stream>>>(buf3, aW2, ab2, alpha_o);

    // ---- CSOC kernel attention -> latent ----
    csoc_kernel<<<dim3(NSEQ), dim3(DMODEL), 0, stream>>>(x, alpha_o, latent_o);

    // ---- ContactPriorHead: h = latent.cW^T+cb; contact = sigmoid(h.h^T) ----
    run_gemm(0, latent_o, DMODEL, cW, DMODEL, cb, buf3, DMODEL,
             NSEQ, DMODEL, DMODEL, stream);
    run_gemm(3, buf3, DMODEL, buf3, DMODEL, nullptr, contact_o, NSEQ,
             NSEQ, NSEQ, DMODEL, stream);

    // ---- DistogramHead: decompose concat-linear into U[i]+V[j] ----
    run_gemm(0, latent_o, DMODEL, dW1, 2 * DMODEL, db1, Ub, DMODEL,
             NSEQ, DMODEL, DMODEL, stream);
    run_gemm(0, latent_o, DMODEL, dW1 + DMODEL, 2 * DMODEL, nullptr, Vb, DMODEL,
             NSEQ, DMODEL, DMODEL, stream);
    disto_kernel<<<dim3(NSEQ, NSEQ / 16), dim3(128), 0, stream>>>(
        Ub, Vb, W2h, db2, disto_o);

    // ---- TorsionGenerator ----
    run_gemm(2, latent_o, DMODEL, tW1, DMODEL, tb1, buf3, DMODEL,
             NSEQ, DMODEL, DMODEL, stream);
    torsion_kernel<<<dim3(NSEQ), dim3(32), 0, stream>>>(buf3, tW2, tb2, phi_o, psi_o);

    // ---- DiffusionCoordinateInitializer ----
    coords_kernel<<<dim3(NSEQ), dim3(32), 0, stream>>>(latent_o, pW, pb, noise, c_o);
}
